// BaseVectorQuantizer_30150670418589
// MI455X (gfx1250) — matrix operand order repaired and checked by reading, not compiled, not run on hardware
//
#include <hip/hip_runtime.h>

typedef __attribute__((ext_vector_type(16))) __bf16 v16bf;
typedef __attribute__((ext_vector_type(8)))  float  v8f;
typedef __attribute__((ext_vector_type(2)))  float  v2f;

#define NTOK   16384     // B*T
#define EDIM   512
#define CDIM   8
#define KCODES 8192
#define LN_EPS 1e-5f

union FragBF  { uint4 q[2]; v16bf v; };
union HalfRow { uint4 q[2]; __bf16 h[16]; };
union Pack8   { __bf16 h[8]; uint4 q; };

// ---------------------------------------------------------------------------
// fp32 -> bf16 convert (8 elems/thread, b128 in / b128 out)
// ---------------------------------------------------------------------------
__global__ __launch_bounds__(256) void cvt_f32_bf16(
    const float* __restrict__ x, __bf16* __restrict__ y)
{
    const size_t i = (size_t)blockIdx.x * 256 + threadIdx.x;   // 8 elems each
    float4 a = *(const float4*)(x + i * 8);
    float4 b = *(const float4*)(x + i * 8 + 4);
    Pack8 o;
    o.h[0]=(__bf16)a.x; o.h[1]=(__bf16)a.y; o.h[2]=(__bf16)a.z; o.h[3]=(__bf16)a.w;
    o.h[4]=(__bf16)b.x; o.h[5]=(__bf16)b.y; o.h[6]=(__bf16)b.z; o.h[7]=(__bf16)b.w;
    *(uint4*)(y + i * 8) = o.q;
}

// ---------------------------------------------------------------------------
// W (fp32, [k][n] 512x512) -> Wt (bf16, [n][k]); coalesced writes, L2 reads
// ---------------------------------------------------------------------------
__global__ __launch_bounds__(256) void transpose_cvt_512(
    const float* __restrict__ W, __bf16* __restrict__ Wt)
{
    const size_t o = (size_t)blockIdx.x * 256 + threadIdx.x;   // o = n*512 + k
    const int n = (int)(o >> 9), k = (int)(o & 511);
    Wt[o] = (__bf16)W[(size_t)k * EDIM + n];
}

// ---------------------------------------------------------------------------
// Async-stage a 128x32 bf16 tile (row stride EDIM) into LDS (row stride 72).
// One global_load_async_to_lds_b128 per 16B per lane; tracked on ASYNCcnt.
// ---------------------------------------------------------------------------
__device__ __forceinline__ void stage_tile_async(
    const __bf16* __restrict__ gbase, __bf16* sbase, int tid)
{
    #pragma unroll
    for (int j = 0; j < 2; ++j) {
        const int g  = tid + 256 * j;        // 512 groups of 8 bf16
        const int r  = g >> 2;
        const int c8 = (g & 3) * 8;
        const __bf16* gp = gbase + (size_t)r * EDIM + c8;
        const unsigned loff = (unsigned)(uintptr_t)(sbase + r * 72 + c8);
        asm volatile("global_load_async_to_lds_b128 %0, %1, off"
                     :: "v"(loff), "v"(gp) : "memory");
    }
}

// ---------------------------------------------------------------------------
// Tiled WMMA GEMM: C[NTOK x 512] = A[NTOK x 512](bf16) x Bt[n][k](bf16) +bias
// Block tile 128x128, 8 wave32, each wave 64x32 = 4x2 WMMA frags, BK=32.
// Double-buffered LDS (72KB); tiles staged via async global->LDS DMA.
// ---------------------------------------------------------------------------
template<bool RELU, bool OUT_BF16>
__global__ __launch_bounds__(256) void gemm512_wmma(
    const __bf16* __restrict__ A, const __bf16* __restrict__ Bt,
    const float* __restrict__ bias, void* __restrict__ Out)
{
    __shared__ __align__(16) __bf16 sA[2 * 128 * 72];
    __shared__ __align__(16) __bf16 sB[2 * 128 * 72];

    const int tid    = threadIdx.x;
    const int lane   = tid & 31;
    const int wave   = tid >> 5;
    const int lane16 = lane & 15;
    const int hi     = lane >> 4;
    const int wm     = wave & 1;   // 2 row groups of 64
    const int wn     = wave >> 1;  // 4 col groups of 32
    const int rowBase = blockIdx.y * 128;
    const int colBase = blockIdx.x * 128;

    const __bf16* Abase = A  + (size_t)rowBase * EDIM;
    const __bf16* Bbase = Bt + (size_t)colBase * EDIM;

    v8f acc[4][2];
    const v8f vzero = {0.f,0.f,0.f,0.f,0.f,0.f,0.f,0.f};
    #pragma unroll
    for (int i = 0; i < 4; ++i)
        #pragma unroll
        for (int j = 0; j < 2; ++j) acc[i][j] = vzero;

    // prologue: stage k-tile 0 into buffer 0
    stage_tile_async(Abase, sA, tid);
    stage_tile_async(Bbase, sB, tid);

    const int kIters = EDIM / 32;
    int buf = 0;
    for (int it = 0; it < kIters; ++it) {
        // staging of current buffer complete across the block
        asm volatile("s_wait_asynccnt 0x0" ::: "memory");
        __syncthreads();

        // overlap: issue async staging of the next k-tile into the other buffer
        if (it + 1 < kIters) {
            const int kn = (it + 1) * 32;
            __builtin_prefetch(Abase + (size_t)(tid >> 1) * EDIM + kn, 0, 0);
            stage_tile_async(Abase + kn, sA + (buf ^ 1) * (128 * 72), tid);
            stage_tile_async(Bbase + kn, sB + (buf ^ 1) * (128 * 72), tid);
        }

        // fragments from current buffer (ISA 16-bit A/B lane layout) + WMMA
        const __bf16* pA = sA + buf * (128 * 72);
        const __bf16* pB = sB + buf * (128 * 72);
        FragBF a[4], b[2];
        #pragma unroll
        for (int i = 0; i < 4; ++i) {
            const int r = wm * 64 + i * 16 + lane16;
            a[i].q[0] = *(const uint4*)(pA + r * 72 + hi * 8);        // K 0-7 / 8-15
            a[i].q[1] = *(const uint4*)(pA + r * 72 + 16 + hi * 8);   // K 16-23 / 24-31
        }
        #pragma unroll
        for (int j = 0; j < 2; ++j) {
            const int n = wn * 32 + j * 16 + lane16;
            b[j].q[0] = *(const uint4*)(pB + n * 72 + hi * 8);        // K 0-7 / 8-15
            b[j].q[1] = *(const uint4*)(pB + n * 72 + 16 + hi * 8);   // K 16-23 / 24-31
        }
        #pragma unroll
        for (int i = 0; i < 4; ++i)
            #pragma unroll
            for (int j = 0; j < 2; ++j)
                acc[i][j] = __builtin_amdgcn_wmma_f32_16x16x32_bf16(
                    false, a[i].v, false, b[j].v, (short)0, acc[i][j], false, false);
        buf ^= 1;
    }

    // ---- epilogue: bias (+ReLU), write bf16 or fp32 ----
    #pragma unroll
    for (int i = 0; i < 4; ++i)
        #pragma unroll
        for (int j = 0; j < 2; ++j) {
            const int n = colBase + wn * 32 + j * 16 + lane16;
            const float bs = bias[n];
            #pragma unroll
            for (int v = 0; v < 8; ++v) {
                const int m = rowBase + wm * 64 + i * 16 + hi * 8 + v;
                float val = acc[i][j][v] + bs;
                if (RELU) val = fmaxf(val, 0.0f);
                if (OUT_BF16) ((__bf16*)Out)[(size_t)m * EDIM + n] = (__bf16)val;
                else          ((float*) Out)[(size_t)m * EDIM + n] = val;
            }
        }
}

// ---------------------------------------------------------------------------
// h (bf16, N x 512) @ W2_in (512 x 8) + b2, layernorm over C=8 -> zf fp32.
// One wave32 per row; lane handles 16 k's; shfl_xor reduction.
// ---------------------------------------------------------------------------
__global__ __launch_bounds__(256) void proj_in_c8_ln(
    const __bf16* __restrict__ h, const float* __restrict__ W2,
    const float* __restrict__ b2, const float* __restrict__ g,
    const float* __restrict__ bv, float* __restrict__ zf)
{
    const int tid = threadIdx.x, lane = tid & 31, wave = tid >> 5;
    const int row = blockIdx.x * 8 + wave;

    HalfRow hr;
    hr.q[0] = *(const uint4*)(h + (size_t)row * EDIM + lane * 16);
    hr.q[1] = *(const uint4*)(h + (size_t)row * EDIM + lane * 16 + 8);

    float acc[8] = {0.f,0.f,0.f,0.f,0.f,0.f,0.f,0.f};
    #pragma unroll
    for (int k = 0; k < 16; ++k) {
        float hv = (float)hr.h[k];
        const float* w = W2 + (size_t)(lane * 16 + k) * CDIM;
        #pragma unroll
        for (int c = 0; c < 8; ++c) acc[c] += hv * w[c];
    }
    #pragma unroll
    for (int c = 0; c < 8; ++c) {
        float v = acc[c];
        #pragma unroll
        for (int off = 16; off >= 1; off >>= 1) v += __shfl_xor(v, off, 32);
        acc[c] = v + b2[c];
    }
    if (lane == 0) {
        float mu = 0.f;
        #pragma unroll
        for (int c = 0; c < 8; ++c) mu += acc[c];
        mu *= 0.125f;
        float var = 0.f;
        #pragma unroll
        for (int c = 0; c < 8; ++c) { float d = acc[c] - mu; var += d * d; }
        var *= 0.125f;
        float sc = rsqrtf(var + LN_EPS);
        float o[8];
        #pragma unroll
        for (int c = 0; c < 8; ++c) o[c] = (acc[c] - mu) * sc * g[c] + bv[c];
        *(float4*)(zf + (size_t)row * CDIM)     = make_float4(o[0], o[1], o[2], o[3]);
        *(float4*)(zf + (size_t)row * CDIM + 4) = make_float4(o[4], o[5], o[6], o[7]);
    }
}

// ---------------------------------------------------------------------------
// VQ via fp32 WMMA: d = ||e||^2 - 2 z.e computed as (-2 zf) x emb^T + C,
// C preloaded with code norms; two chained V_WMMA_F32_16X16X4_F32 per
// 16-row x 16-code tile. One wave = 16 rows x all 8192 codes (LDS-chunked).
// Then per-row argmin reduction (shfl, low-index tie-break), one-hot scatter.
// ---------------------------------------------------------------------------
#define VQ_CHUNK 2048
__global__ __launch_bounds__(256) void vq_wmma_argmin(
    const float* __restrict__ zf, const float* __restrict__ emb,
    float* __restrict__ idx_out, float* __restrict__ enc_out,
    float* __restrict__ quant_out)
{
    __shared__ __align__(16) float sE[VQ_CHUNK * 8];   // 64 KB codebook chunk
    __shared__ float sN[VQ_CHUNK];                     // 8 KB norms
    __shared__ int   sIdx[128];                        // per-row winners

    const int tid = threadIdx.x, lane = tid & 31, wave = tid >> 5;
    const int lane16 = lane & 15, hi = lane >> 4;
    const int r0 = (blockIdx.x * 8 + wave) * 16;       // 16 rows per wave

    // A fragments (16x4 fp32 layout: lanes 0-15 rows, K = 2*hi + {0,1}):
    // fold the -2 scale of the cross term into A.
    const float* zr = zf + (size_t)(r0 + lane16) * CDIM;
    v2f a1, a2;
    a1[0] = -2.0f * zr[2 * hi + 0];
    a1[1] = -2.0f * zr[2 * hi + 1];
    a2[0] = -2.0f * zr[4 + 2 * hi + 0];
    a2[1] = -2.0f * zr[4 + 2 * hi + 1];

    float bestV[8];
    int   bestI[8];
    #pragma unroll
    for (int v = 0; v < 8; ++v) { bestV[v] = 3.4e38f; bestI[v] = 0; }

    for (int c0 = 0; c0 < KCODES; c0 += VQ_CHUNK) {
        __syncthreads();
        #pragma unroll
        for (int j = 0; j < 16; ++j) {                 // stage chunk (4096 b128)
            int g = tid + 256 * j;
            *(float4*)(sE + g * 4) = *(const float4*)(emb + (size_t)c0 * 8 + g * 4);
        }
        __syncthreads();
        #pragma unroll
        for (int j = 0; j < 8; ++j) {                  // code norms ||e||^2
            int k = tid + 256 * j;
            const float* e = sE + k * 8;
            float nn = 0.f;
            #pragma unroll
            for (int c = 0; c < 8; ++c) nn += e[c] * e[c];
            sN[k] = nn;
        }
        __syncthreads();

        for (int t = 0; t < VQ_CHUNK / 16; ++t) {      // 128 tiles of 16 codes
            const int code = t * 16 + lane16;
            // B fragments (4x16 fp32: lanes 0-15 cols, K = 2*hi + {0,1})
            v2f b1 = *(const v2f*)(sE + code * 8 + 2 * hi);
            v2f b2 = *(const v2f*)(sE + code * 8 + 4 + 2 * hi);
            const float nrm = sN[code];
            v8f d = {nrm, nrm, nrm, nrm, nrm, nrm, nrm, nrm};
            d = __builtin_amdgcn_wmma_f32_16x16x4_f32(
                    false, a1, false, b1, (short)0, d, false, false);
            d = __builtin_amdgcn_wmma_f32_16x16x4_f32(
                    false, a2, false, b2, (short)0, d, false, false);
            const int gidx = c0 + code;
            #pragma unroll
            for (int v = 0; v < 8; ++v) {
                if (d[v] < bestV[v]) { bestV[v] = d[v]; bestI[v] = gidx; }
            }
        }
    }

    // reduce across the 16 lanes of each half-wave (lanes0-15: M=v; 16-31: M=v+8)
    #pragma unroll
    for (int off = 1; off < 16; off <<= 1) {
        #pragma unroll
        for (int v = 0; v < 8; ++v) {
            float ov = __shfl_xor(bestV[v], off, 32);
            int   oi = __shfl_xor(bestI[v], off, 32);
            if (ov < bestV[v] || (ov == bestV[v] && oi < bestI[v])) {
                bestV[v] = ov; bestI[v] = oi;
            }
        }
    }
    if (lane16 == 0) {
        #pragma unroll
        for (int v = 0; v < 8; ++v)
            sIdx[wave * 16 + hi * 8 + v] = bestI[v];
    }
    __syncthreads();

    // zero this block's encodings slice (128 rows x 8192 fp32 = 4MB), coalesced
    const float4 zero4 = make_float4(0.f, 0.f, 0.f, 0.f);
    float* base = enc_out + (size_t)blockIdx.x * 128 * KCODES;
    for (int j = 0; j < (128 * KCODES / 4) / 256; ++j)
        *(float4*)(base + (size_t)(j * 256 + tid) * 4) = zero4;
    __threadfence_block();
    __syncthreads();

    // per-row outputs: idx (as float), quantized gather, one-hot scatter
    if (tid < 128) {
        const int row = blockIdx.x * 128 + tid;
        const int bi  = sIdx[tid];
        idx_out[row] = (float)bi;
        const float* eb = emb + (size_t)bi * 8;
        *(float4*)(quant_out + (size_t)row * 8)     = *(const float4*)eb;
        *(float4*)(quant_out + (size_t)row * 8 + 4) = *(const float4*)(eb + 4);
        enc_out[(size_t)row * KCODES + bi] = 1.0f;
    }
}

// ---------------------------------------------------------------------------
// quantized (N x 8) @ W1_out (8 x 512) + b1, ReLU -> q1 bf16. Wave per row.
// ---------------------------------------------------------------------------
__global__ __launch_bounds__(256) void proj_out_e512(
    const float* __restrict__ quant, const float* __restrict__ W1,
    const float* __restrict__ b1, __bf16* __restrict__ q1)
{
    const int tid = threadIdx.x, lane = tid & 31, wave = tid >> 5;
    const int row = blockIdx.x * 8 + wave;

    float qv[8];
    {
        float4 a = *(const float4*)(quant + (size_t)row * 8);
        float4 b = *(const float4*)(quant + (size_t)row * 8 + 4);
        qv[0]=a.x; qv[1]=a.y; qv[2]=a.z; qv[3]=a.w;
        qv[4]=b.x; qv[5]=b.y; qv[6]=b.z; qv[7]=b.w;
    }
    const int e0 = lane * 16;
    float acc[16];
    #pragma unroll
    for (int t = 0; t < 16; ++t) acc[t] = b1[e0 + t];
    #pragma unroll
    for (int c = 0; c < 8; ++c) {
        const float* w = W1 + (size_t)c * EDIM + e0;
        #pragma unroll
        for (int t = 0; t < 16; ++t) acc[t] += qv[c] * w[t];
    }
    HalfRow o;
    #pragma unroll
    for (int t = 0; t < 16; ++t) o.h[t] = (__bf16)fmaxf(acc[t], 0.0f);
    *(uint4*)(q1 + (size_t)row * EDIM + e0)     = o.q[0];
    *(uint4*)(q1 + (size_t)row * EDIM + e0 + 8) = o.q[1];
}

// ---------------------------------------------------------------------------
// LayerNorm over E=512 per row (fp32 in -> fp32 out). Wave per row.
// ---------------------------------------------------------------------------
__global__ __launch_bounds__(256) void ln512_kernel(
    const float* __restrict__ x, const float* __restrict__ g,
    const float* __restrict__ b, float* __restrict__ out)
{
    const int tid = threadIdx.x, lane = tid & 31, wave = tid >> 5;
    const int row = blockIdx.x * 8 + wave;
    const int e0  = lane * 16;
    const float* xr = x + (size_t)row * EDIM + e0;

    float v[16];
    #pragma unroll
    for (int j = 0; j < 4; ++j) {
        float4 d = *(const float4*)(xr + j * 4);
        v[j*4+0]=d.x; v[j*4+1]=d.y; v[j*4+2]=d.z; v[j*4+3]=d.w;
    }
    float s = 0.f, s2 = 0.f;
    #pragma unroll
    for (int t = 0; t < 16; ++t) { s += v[t]; s2 += v[t] * v[t]; }
    #pragma unroll
    for (int off = 16; off >= 1; off >>= 1) {
        s  += __shfl_xor(s,  off, 32);
        s2 += __shfl_xor(s2, off, 32);
    }
    const float mu  = s  * (1.0f / EDIM);
    const float var = s2 * (1.0f / EDIM) - mu * mu;
    const float sc  = rsqrtf(var + LN_EPS);

    float* orow = out + (size_t)row * EDIM + e0;
    #pragma unroll
    for (int j = 0; j < 4; ++j) {
        float4 d;
        d.x = (v[j*4+0] - mu) * sc * g[e0 + j*4+0] + b[e0 + j*4+0];
        d.y = (v[j*4+1] - mu) * sc * g[e0 + j*4+1] + b[e0 + j*4+1];
        d.z = (v[j*4+2] - mu) * sc * g[e0 + j*4+2] + b[e0 + j*4+2];
        d.w = (v[j*4+3] - mu) * sc * g[e0 + j*4+3] + b[e0 + j*4+3];
        *(float4*)(orow + j * 4) = d;
    }
}

// ---------------------------------------------------------------------------
extern "C" void kernel_launch(void* const* d_in, const int* in_sizes, int n_in,
                              void* d_out, int out_size, void* d_ws, size_t ws_size,
                              hipStream_t stream)
{
    (void)in_sizes; (void)n_in; (void)out_size; (void)ws_size;

    const float* features = (const float*)d_in[0];
    const float* W1_in    = (const float*)d_in[1];
    const float* b1_in    = (const float*)d_in[2];
    const float* W2_in    = (const float*)d_in[3];
    const float* b2_in    = (const float*)d_in[4];
    const float* ln_in_g  = (const float*)d_in[5];
    const float* ln_in_b  = (const float*)d_in[6];
    const float* emb      = (const float*)d_in[7];
    const float* W1_out   = (const float*)d_in[8];
    const float* b1_out   = (const float*)d_in[9];
    const float* W2_out   = (const float*)d_in[10];
    const float* b2_out   = (const float*)d_in[11];
    const float* ln_out_g = (const float*)d_in[12];
    const float* ln_out_b = (const float*)d_in[13];

    // d_out layout: q (N x 512) | idx (N) | encodings (N x 8192), all fp32
    float* q_out   = (float*)d_out;
    float* idx_out = q_out + (size_t)NTOK * EDIM;
    float* enc_out = idx_out + NTOK;

    // workspace layout
    char*   ws    = (char*)d_ws;
    __bf16* featb = (__bf16*)ws;  ws += (size_t)NTOK * EDIM * 2;  // 16 MB
    __bf16* W1t   = (__bf16*)ws;  ws += (size_t)EDIM * EDIM * 2;  // 512 KB
    __bf16* W2t   = (__bf16*)ws;  ws += (size_t)EDIM * EDIM * 2;  // 512 KB
    __bf16* h     = (__bf16*)ws;  ws += (size_t)NTOK * EDIM * 2;  // 16 MB
    float*  zf    = (float*)ws;   ws += (size_t)NTOK * CDIM * 4;  // 512 KB
    float*  quant = (float*)ws;   ws += (size_t)NTOK * CDIM * 4;  // 512 KB
    __bf16* q1    = (__bf16*)ws;  ws += (size_t)NTOK * EDIM * 2;  // 16 MB
    float*  q2    = (float*)ws;                                   // 32 MB

    const dim3 gemm_grid(EDIM / 128, NTOK / 128);   // (4, 128)

    // 0) precision prep: activations fp32->bf16, weights fp32->bf16 transposed
    cvt_f32_bf16<<<(NTOK * EDIM / 8) / 256, 256, 0, stream>>>(features, featb);
    transpose_cvt_512<<<(EDIM * EDIM) / 256, 256, 0, stream>>>(W1_in, W1t);
    transpose_cvt_512<<<(EDIM * EDIM) / 256, 256, 0, stream>>>(W2_out, W2t);

    // 1) h = relu(features @ W1_in + b1)            [WMMA bf16, async LDS]
    gemm512_wmma<true, true><<<gemm_grid, 256, 0, stream>>>(featb, W1t, b1_in, h);
    // 2) z = LN_8(h @ W2_in + b2)
    proj_in_c8_ln<<<NTOK / 8, 256, 0, stream>>>(h, W2_in, b2_in, ln_in_g, ln_in_b, zf);
    // 3) idx / quantized / encodings               [WMMA fp32 16x16x4]
    vq_wmma_argmin<<<NTOK / 128, 256, 0, stream>>>(zf, emb, idx_out, enc_out, quant);
    // 4) q1 = relu(quant @ W1_out + b1_out)
    proj_out_e512<<<NTOK / 8, 256, 0, stream>>>(quant, W1_out, b1_out, q1);
    // 5) q2 = q1 @ W2_out + b2_out                  [WMMA bf16, async LDS]
    gemm512_wmma<false, false><<<gemm_grid, 256, 0, stream>>>(q1, W2t, b2_out, q2);
    // 6) q = LN_512(q2)
    ln512_kernel<<<NTOK / 8, 256, 0, stream>>>(q2, ln_out_g, ln_out_b, q_out);
}
